// LocalGlobalGNN_4672924418435
// MI455X (gfx1250) — matrix operand order repaired and checked by reading, not compiled, not run on hardware
//
#include <hip/hip_runtime.h>

// ---------------------------------------------------------------------------
// CDNA5 (gfx1250) LocalGlobalGNN forward:
//   5x SAGEConv (scatter-mean via f32 atomics + dual bf16-WMMA GEMM)
//   + fused concat-MLP1 (dual WMMA GEMM) + fused BN+ReLU MLP2 (WMMA GEMM)
// GEMM: one wave computes a 16x64 output tile (4x v_wmma_f32_16x16x32_bf16
// per A-fragment), fp32 accumulate, weights pre-transposed to bf16.
// ---------------------------------------------------------------------------

typedef __attribute__((ext_vector_type(16))) __bf16 v16bf;
typedef __attribute__((ext_vector_type(8)))  __bf16 v8bf;
typedef __attribute__((ext_vector_type(8)))  float  v8f;

__device__ __forceinline__ __bf16 f2bf(float f) { return (__bf16)f; }

// ------------------------------ utility kernels ----------------------------

__global__ void k_zero(float* __restrict__ p, long long n) {
  long long i = (long long)blockIdx.x * blockDim.x + threadIdx.x;
  long long stride = (long long)gridDim.x * blockDim.x;
  for (; i < n; i += stride) p[i] = 0.0f;
}

__global__ void k_degree(const int* __restrict__ dst, float* __restrict__ deg, int E) {
  int i = blockIdx.x * blockDim.x + threadIdx.x;
  if (i < E) atomicAdd(&deg[dst[i]], 1.0f);
}

__global__ void k_invdeg(float* __restrict__ deg, int n) {
  int i = blockIdx.x * blockDim.x + threadIdx.x;
  if (i < n) deg[i] = 1.0f / fmaxf(deg[i], 1.0f);
}

// agg[dst] += h[src] over feature dim d (32 lanes stride the feature dim)
__global__ void k_scatter_add(const float* __restrict__ h, const int* __restrict__ src,
                              const int* __restrict__ dst, float* __restrict__ agg,
                              int E, int d) {
  int e = blockIdx.x * blockDim.y + threadIdx.y;
  if (e >= E) return;
  const float* hs = h + (size_t)src[e] * d;
  float* ad = agg + (size_t)dst[e] * d;
  for (int c = threadIdx.x; c < d; c += 32)
    atomicAdd(&ad[c], hs[c]);
}

// W [K, Hc] f32 row-major  ->  Wt [Hc, K] bf16 row-major (transpose + convert)
__global__ void k_wprep(const float* __restrict__ W, __bf16* __restrict__ Wt,
                        int K, int Hc) {
  int i = blockIdx.x * blockDim.x + threadIdx.x;
  if (i >= K * Hc) return;
  int k = i / Hc, h = i - k * Hc;
  Wt[(size_t)h * K + k] = f2bf(W[i]);
}

// BatchNorm stats -> per-column affine: s = gamma*rsqrt(var+eps), t = beta - mu*s
__global__ void k_bn_stats(const float* __restrict__ x, const float* __restrict__ gamma,
                           const float* __restrict__ beta, float* __restrict__ s,
                           float* __restrict__ t, int n, int d) {
  __shared__ float sh_sum[256];
  __shared__ float sh_sq[256];
  int j = blockIdx.x;
  float sum = 0.0f, sq = 0.0f;
  for (int r = threadIdx.x; r < n; r += blockDim.x) {
    float v = x[(size_t)r * d + j];
    sum += v; sq += v * v;
  }
  sh_sum[threadIdx.x] = sum; sh_sq[threadIdx.x] = sq;
  __syncthreads();
  for (int w = blockDim.x >> 1; w > 0; w >>= 1) {
    if (threadIdx.x < (unsigned)w) {
      sh_sum[threadIdx.x] += sh_sum[threadIdx.x + w];
      sh_sq[threadIdx.x]  += sh_sq[threadIdx.x + w];
    }
    __syncthreads();
  }
  if (threadIdx.x == 0) {
    float mu  = sh_sum[0] / n;
    float var = sh_sq[0] / n - mu * mu;
    float sc  = gamma[j] * rsqrtf(var + 1e-5f);
    s[j] = sc;
    t[j] = beta[j] - mu * sc;
  }
}

// ------------------------------ WMMA GEMM ----------------------------------
// A fragment (16x32 bf16, ISA 7.12.2): lane L: M = L&15.
//   half = L>>4:  elements[0..7]  = K = k0 + half*8      + (0..7)
//                 elements[8..15] = K = k0 + 16 + half*8 + (0..7)
__device__ __forceinline__ v16bf load_a_frag(const float* __restrict__ p, float scale) {
  const float4* p4 = (const float4*)p;
  float4 f0 = p4[0], f1 = p4[1], g0 = p4[4], g1 = p4[5];
  v16bf a;
  a[0]  = f2bf(f0.x * scale); a[1]  = f2bf(f0.y * scale);
  a[2]  = f2bf(f0.z * scale); a[3]  = f2bf(f0.w * scale);
  a[4]  = f2bf(f1.x * scale); a[5]  = f2bf(f1.y * scale);
  a[6]  = f2bf(f1.z * scale); a[7]  = f2bf(f1.w * scale);
  a[8]  = f2bf(g0.x * scale); a[9]  = f2bf(g0.y * scale);
  a[10] = f2bf(g0.z * scale); a[11] = f2bf(g0.w * scale);
  a[12] = f2bf(g1.x * scale); a[13] = f2bf(g1.y * scale);
  a[14] = f2bf(g1.z * scale); a[15] = f2bf(g1.w * scale);
  return a;
}

// B fragment (32x16 bf16): lane L: N = L&15, K = k0 + (L>>4)*8 + (0..7) and
// k0 + 16 + (L>>4)*8 + (0..7), from Wt[N, :] (transposed weight). Two 16B loads.
__device__ __forceinline__ v16bf load_b_frag(const __bf16* __restrict__ p) {
  const v8bf* q = (const v8bf*)p;
  v8bf b0 = q[0], b1 = q[2];
  v16bf b;
#pragma unroll
  for (int i = 0; i < 8; ++i) { b[i] = b0[i]; b[8 + i] = b1[i]; }
  return b;
}

// MODE 0: out = A0 @ Wt0^T + bias
// MODE 1: out = A0 @ Wt0^T + (A1 * invdeg?) @ Wt1^T + bias   (SAGE / concat-MLP1)
// MODE 2: out = relu(A0 * s[k] + t[k]) @ Wt0^T + bias        (fused BatchNorm)
// One wave -> 16 rows x 64 cols (4 WMMA column subtiles sharing each A frag).
template <int MODE>
__global__ __launch_bounds__(128)
void k_wmma_gemm(const float* __restrict__ A0, const float* __restrict__ A1,
                 const float* __restrict__ invdeg,
                 const __bf16* __restrict__ Wt0, const __bf16* __restrict__ Wt1,
                 const float* __restrict__ bias,
                 const float* __restrict__ aff_s, const float* __restrict__ aff_t,
                 float* __restrict__ out, int rows, int K, int Hout, int relu) {
  const int lane = threadIdx.x;        // 0..31 (wave32)
  const int hi   = lane >> 4;
  const int l16  = lane & 15;
  const int mtiles = rows >> 4;
  const int tile_m = blockIdx.x * blockDim.y + threadIdx.y;
  if (tile_m >= mtiles) return;        // wave-uniform: EXEC stays all-ones
  const int n0 = blockIdx.y * 64;      // 4 x 16 column subtiles

  const int row = tile_m * 16 + l16;   // A row this lane supplies (both halves)
  const size_t a_off = (size_t)row * K;
  size_t boff[4];
#pragma unroll
  for (int j = 0; j < 4; ++j) boff[j] = (size_t)(n0 + j * 16 + l16) * K;

  float scale1 = 1.0f;
  if (MODE == 1) scale1 = invdeg ? invdeg[row] : 1.0f;

  v8f acc[4] = {};
  for (int k0 = 0; k0 < K; k0 += 32) {
    v16bf a;
    if (MODE == 2) {  // fused BatchNorm affine + ReLU on A0
      const float* pa = A0 + a_off + k0 + hi * 8;
      const float* ps = aff_s + k0 + hi * 8;
      const float* pt = aff_t + k0 + hi * 8;
#pragma unroll
      for (int i = 0; i < 8; ++i) {
        a[i]     = f2bf(fmaxf(fmaf(pa[i],      ps[i],      pt[i]),      0.0f));
        a[8 + i] = f2bf(fmaxf(fmaf(pa[16 + i], ps[16 + i], pt[16 + i]), 0.0f));
      }
    } else {
      a = load_a_frag(A0 + a_off + k0 + hi * 8, 1.0f);
    }
    __builtin_prefetch(A0 + a_off + k0 + 32, 0, 1);  // speculative: safe past tail
#pragma unroll
    for (int j = 0; j < 4; ++j) {
      v16bf b = load_b_frag(Wt0 + boff[j] + k0 + hi * 8);
      acc[j] = __builtin_amdgcn_wmma_f32_16x16x32_bf16(false, a, false, b,
                                                       (short)0, acc[j], false, false);
    }
    if (MODE == 1) {
      v16bf a1 = load_a_frag(A1 + a_off + k0 + hi * 8, scale1);
#pragma unroll
      for (int j = 0; j < 4; ++j) {
        v16bf b = load_b_frag(Wt1 + boff[j] + k0 + hi * 8);
        acc[j] = __builtin_amdgcn_wmma_f32_16x16x32_bf16(false, a1, false, b,
                                                         (short)0, acc[j], false, false);
      }
    }
  }

  // D layout: lane half selects rows v+8*hi, column = l16 (per subtile)
#pragma unroll
  for (int j = 0; j < 4; ++j) {
    const int col = n0 + j * 16 + l16;
    const float bv = bias ? bias[col] : 0.0f;
    float* op = out + (size_t)(tile_m * 16 + hi * 8) * Hout + col;
#pragma unroll
    for (int v = 0; v < 8; ++v) {
      float o = acc[j][v] + bv;
      if (relu) o = fmaxf(o, 0.0f);
      op[(size_t)v * Hout] = o;
    }
  }
}

// ------------------------------ host driver --------------------------------

extern "C" void kernel_launch(void* const* d_in, const int* in_sizes, int n_in,
                              void* d_out, int out_size, void* d_ws, size_t ws_size,
                              hipStream_t stream) {
  (void)n_in; (void)out_size; (void)ws_size;
  const int IN = 128, H = 256, C = 64, MLPH = 128;
  const int N  = in_sizes[0] / IN;
  const int EG = in_sizes[1];
  const int EK = in_sizes[3];

  const float* feat   = (const float*)d_in[0];
  const int*   g_src  = (const int*)d_in[1];
  const int*   g_dst  = (const int*)d_in[2];
  const int*   k_src  = (const int*)d_in[3];
  const int*   k_dst  = (const int*)d_in[4];
  const float* lws0 = (const float*)d_in[5],  *lwn0 = (const float*)d_in[6],  *lb0 = (const float*)d_in[7];
  const float* lws1 = (const float*)d_in[8],  *lwn1 = (const float*)d_in[9],  *lb1 = (const float*)d_in[10];
  const float* lws2 = (const float*)d_in[11], *lwn2 = (const float*)d_in[12], *lb2 = (const float*)d_in[13];
  const float* gws0 = (const float*)d_in[14], *gwn0 = (const float*)d_in[15], *gb0 = (const float*)d_in[16];
  const float* gws1 = (const float*)d_in[17], *gwn1 = (const float*)d_in[18], *gb1 = (const float*)d_in[19];
  const float* mlp_w1 = (const float*)d_in[20], *mlp_b1 = (const float*)d_in[21];
  const float* bn_g   = (const float*)d_in[22], *bn_b   = (const float*)d_in[23];
  const float* mlp_w2 = (const float*)d_in[24], *mlp_b2 = (const float*)d_in[25];

  // ---- workspace layout ----
  char* ws = (char*)d_ws;
  size_t off = 0;
  auto take = [&](size_t b) { size_t o = off; off += (b + 255) & ~(size_t)255; return o; };
  float*  BA     = (float*)(ws + take((size_t)N * H * 4));
  float*  BB     = (float*)(ws + take((size_t)N * H * 4));
  float*  BAGG   = (float*)(ws + take((size_t)N * H * 4));   // agg scratch; later X1
  float*  BLOCAL = (float*)(ws + take((size_t)N * H * 4));
  float*  idg    = (float*)(ws + take((size_t)N * 4));
  float*  idk    = (float*)(ws + take((size_t)N * 4));
  float*  bn_s   = (float*)(ws + take(MLPH * 4));
  float*  bn_t   = (float*)(ws + take(MLPH * 4));
  __bf16* wl0s = (__bf16*)(ws + take((size_t)IN * H * 2));
  __bf16* wl0n = (__bf16*)(ws + take((size_t)IN * H * 2));
  __bf16* wl1s = (__bf16*)(ws + take((size_t)H * H * 2));
  __bf16* wl1n = (__bf16*)(ws + take((size_t)H * H * 2));
  __bf16* wl2s = (__bf16*)(ws + take((size_t)H * H * 2));
  __bf16* wl2n = (__bf16*)(ws + take((size_t)H * H * 2));
  __bf16* wg0s = (__bf16*)(ws + take((size_t)IN * H * 2));
  __bf16* wg0n = (__bf16*)(ws + take((size_t)IN * H * 2));
  __bf16* wg1s = (__bf16*)(ws + take((size_t)H * H * 2));
  __bf16* wg1n = (__bf16*)(ws + take((size_t)H * H * 2));
  __bf16* w1top = (__bf16*)(ws + take((size_t)H * MLPH * 2));
  __bf16* w1bot = (__bf16*)(ws + take((size_t)H * MLPH * 2));
  __bf16* w2t   = (__bf16*)(ws + take((size_t)MLPH * C * 2));
  float* X1 = BAGG;  // reuse agg scratch for MLP hidden activations

  // ---- degrees -> inverse mean denominators ----
  k_zero<<<1024, 256, 0, stream>>>(idg, (long long)N);
  k_zero<<<1024, 256, 0, stream>>>(idk, (long long)N);
  k_degree<<<(EG + 255) / 256, 256, 0, stream>>>(g_dst, idg, EG);
  k_degree<<<(EK + 255) / 256, 256, 0, stream>>>(k_dst, idk, EK);
  k_invdeg<<<(N + 255) / 256, 256, 0, stream>>>(idg, N);
  k_invdeg<<<(N + 255) / 256, 256, 0, stream>>>(idk, N);

  // ---- weight prep (transpose + fp32 -> bf16) ----
  auto prep = [&](const float* W, __bf16* Wt, int K, int Hc) {
    int tot = K * Hc;
    k_wprep<<<(tot + 255) / 256, 256, 0, stream>>>(W, Wt, K, Hc);
  };
  prep(lws0, wl0s, IN, H); prep(lwn0, wl0n, IN, H);
  prep(lws1, wl1s, H, H);  prep(lwn1, wl1n, H, H);
  prep(lws2, wl2s, H, H);  prep(lwn2, wl2n, H, H);
  prep(gws0, wg0s, IN, H); prep(gwn0, wg0n, IN, H);
  prep(gws1, wg1s, H, H);  prep(gwn1, wg1n, H, H);
  prep(mlp_w1,                    w1top, H, MLPH);  // rows 0..H-1   (local half)
  prep(mlp_w1 + (size_t)H * MLPH, w1bot, H, MLPH);  // rows H..2H-1 (global half)
  prep(mlp_w2, w2t, MLPH, C);

  // ---- one SAGEConv layer: scatter-mean + dual WMMA GEMM ----
  dim3 sb(32, 8);
  dim3 gb(32, 4);
  auto sage = [&](const float* hin, int din, const int* src, const int* dst, int E,
                  const float* invd, __bf16* wS, __bf16* wN, const float* b_,
                  float* out_, int relu) {
    k_zero<<<2048, 256, 0, stream>>>(BAGG, (long long)N * din);
    k_scatter_add<<<(E + 7) / 8, sb, 0, stream>>>(hin, src, dst, BAGG, E, din);
    dim3 gg((N / 16 + 3) / 4, H / 64);
    k_wmma_gemm<1><<<gg, gb, 0, stream>>>(hin, BAGG, invd, wS, wN, b_,
                                          nullptr, nullptr, out_, N, din, H, relu);
  };

  // local GraphSAGE (3 layers)
  sage(feat, IN, g_src, g_dst, EG, idg, wl0s, wl0n, lb0, BA, 1);
  sage(BA,   H,  g_src, g_dst, EG, idg, wl1s, wl1n, lb1, BB, 1);
  sage(BB,   H,  g_src, g_dst, EG, idg, wl2s, wl2n, lb2, BLOCAL, 0);
  // global GraphSAGE (2 layers)
  sage(feat, IN, k_src, k_dst, EK, idk, wg0s, wg0n, gb0, BA, 1);
  sage(BA,   H,  k_src, k_dst, EK, idk, wg1s, wg1n, gb1, BB, 0);

  // MLP1: concat fused as dual GEMM (local@w1_top + global@w1_bot + b1)
  {
    dim3 gg((N / 16 + 3) / 4, MLPH / 64);
    k_wmma_gemm<1><<<gg, gb, 0, stream>>>(BLOCAL, BB, nullptr, w1top, w1bot, mlp_b1,
                                          nullptr, nullptr, X1, N, H, MLPH, 0);
  }
  // BatchNorm stats -> per-column affine (s, t)
  k_bn_stats<<<MLPH, 256, 0, stream>>>(X1, bn_g, bn_b, bn_s, bn_t, N, MLPH);
  // MLP2: fused BN-affine + ReLU on A, GEMM to output
  {
    dim3 gg((N / 16 + 3) / 4, C / 64);
    k_wmma_gemm<2><<<gg, gb, 0, stream>>>(X1, nullptr, nullptr, w2t, nullptr, mlp_b2,
                                          bn_s, bn_t, (float*)d_out, N, MLPH, C, 0);
  }
}
